// GCNNet_79173427134951
// MI455X (gfx1250) — matrix operand order repaired and checked by reading, not compiled, not run on hardware
//
#include <hip/hip_runtime.h>
#include <stdint.h>

// ---------------------------------------------------------------------------
// GAT (3-layer, 4-head) for MI455X / gfx1250, wave32 + WMMA bf16.
//   GEMMs: bf16 inputs, f32 accumulate via v_wmma_f32_16x16x32_bf16.
//          Weight slab staged to LDS with global_load_async_to_lds_b128
//          (ASYNCcnt) and read back as B fragments via ds loads.
//   Edge ops: 4 passes (max / exp-sum / alpha / aggregate), all L2-resident.
// ---------------------------------------------------------------------------

typedef __attribute__((ext_vector_type(16))) __bf16 v16bf;
typedef __attribute__((ext_vector_type(8)))  float  v8f;

__device__ __forceinline__ unsigned short f2bf(float f) {
  unsigned int u = __float_as_uint(f);
  u += 0x7FFFu + ((u >> 16) & 1u);   // round-to-nearest-even
  return (unsigned short)(u >> 16);
}

// monotone mapping float -> u32 so unsigned atomicMax orders like float max
__device__ __forceinline__ unsigned int flip_f32(float f) {
  unsigned int u = __float_as_uint(f);
  return (u & 0x80000000u) ? ~u : (u | 0x80000000u);
}
__device__ __forceinline__ float unflip_f32(unsigned int v) {
  unsigned int u = (v & 0x80000000u) ? (v & 0x7FFFFFFFu) : ~v;
  return __uint_as_float(u);
}

// -------------------------------- utility ----------------------------------

__global__ void fill_u32_kernel(unsigned int* __restrict__ p, unsigned int val,
                                long long n) {
  long long i = (long long)blockIdx.x * blockDim.x + threadIdx.x;
  if (i < n) p[i] = val;
}

__global__ void f32_to_bf16_kernel(const float* __restrict__ in,
                                   unsigned short* __restrict__ out, int n) {
  int i = blockIdx.x * blockDim.x + threadIdx.x;
  if (i < n) out[i] = f2bf(in[i]);
}

// W [K,Nout] f32 row-major  ->  Wt [Nout,K] bf16 (i.e. column-major of W)
__global__ void wt_convert_kernel(const float* __restrict__ W,
                                  unsigned short* __restrict__ Wt,
                                  int K, int Nout) {
  int i = blockIdx.x * blockDim.x + threadIdx.x;
  if (i >= K * Nout) return;
  int k = i % K;
  int n = i / K;
  Wt[n * K + k] = f2bf(W[k * Nout + n]);
}

// ------------------------------ WMMA GEMM ----------------------------------
// C[M,Nout] f32 = A[M,K] bf16 x B[K,Nout] bf16 (B given transposed: Bt[Nout,K])
// Block = 8 waves sharing one 16-column B slab (staged async to LDS);
// each wave computes one 16x16 C tile. K stepped by 32.
__global__ __launch_bounds__(256) void gemm_wmma_bf16(
    const unsigned short* __restrict__ A,   // [M,K]    bf16 bits
    const unsigned short* __restrict__ Bt,  // [Nout,K] bf16 bits
    float* __restrict__ C,                  // [M,Nout]
    int M, int K, int Nout) {
  const int mTiles  = (M + 15) >> 4;
  const int mGroups = (mTiles + 7) >> 3;
  const int mg = blockIdx.x % mGroups;
  const int tn = blockIdx.x / mGroups;

  __shared__ __align__(16) unsigned short smemB[16 * 128];  // 16 cols x K<=128

  // ---- async-stage B slab (16 cols x K bf16) into LDS (ASYNCcnt path) ----
  const int totalElems = 16 * K;            // u16 elements, multiple of 8
  for (int e8 = threadIdx.x * 8; e8 < totalElems; e8 += 256 * 8) {
    int r  = e8 / K;
    int kk = e8 - r * K;
    unsigned lds_off = (unsigned)(uintptr_t)(smemB + e8);  // low32 = LDS addr
    const unsigned short* g = Bt + (size_t)(tn * 16 + r) * K + kk;
    asm volatile("global_load_async_to_lds_b128 %0, %1, off"
                 :: "v"(lds_off), "v"(g) : "memory");
  }
  asm volatile("s_wait_asynccnt 0x0" ::: "memory");
  __syncthreads();

  const int wv   = threadIdx.x >> 5;
  const int tm   = mg * 8 + wv;
  if (tm >= mTiles) return;                 // wave-uniform exit, EXEC stays full
  const int lane = threadIdx.x & 31;
  const int l16  = lane & 15;
  const int hi   = lane >> 4;

  int row = tm * 16 + l16;
  if (row >= M) row = M - 1;                // clamp keeps all lanes active
  const unsigned short* Arow  = A + (size_t)row * K;
  const unsigned short* BrowL = smemB + l16 * K;   // LDS (ds loads)

  v8f acc = {};
  for (int k0 = 0; k0 < K; k0 += 32) {
    union { unsigned int u[8]; v16bf v; } a, b;
#pragma unroll
    for (int v = 0; v < 8; ++v) {
      // 16-bit A-matrix 16x32 layout: lanes0-15 K=2v(+b)+{0|16}, lanes16-31 +8
      int ka = ((v < 4) ? (2 * v) : (2 * v + 8)) + hi * 8 + k0;
      a.u[v] = *(const unsigned int*)(Arow + ka);
      // B-matrix 32x16: same K order as A (lanes0-15 K=2v(+b)+{0|16}, lanes16-31 +8)
      int kb = ((v < 4) ? (2 * v) : (2 * v + 8)) + hi * 8 + k0;
      b.u[v] = *(const unsigned int*)(BrowL + kb);
    }
    acc = __builtin_amdgcn_wmma_f32_16x16x32_bf16(
        /*neg_a=*/false, a.v, /*neg_b=*/false, b.v,
        /*c_mod=*/(short)0, acc, /*reuse_a=*/false, /*reuse_b=*/false);
  }
  const int col = tn * 16 + l16;
#pragma unroll
  for (int r = 0; r < 8; ++r) {
    int orow = tm * 16 + hi * 8 + r;        // C/D layout: VGPR r -> row r (+8 hi)
    if (orow < M) C[(size_t)orow * Nout + col] = acc[r];
  }
}

// --------------------------- attention scores ------------------------------
// ES[n,h] = dot(Hm[n,h,:], a_s[h,:]);  ED likewise. One thread per (n,h).
__global__ void att_scores_kernel(const float* __restrict__ Hm,
                                  const float* __restrict__ as_,
                                  const float* __restrict__ ad_,
                                  float* __restrict__ ES, float* __restrict__ ED,
                                  int N, int H, int D) {
  int i = blockIdx.x * blockDim.x + threadIdx.x;
  if (i >= N * H) return;
  int n = i / H, hh = i - n * H;
  int HD = H * D;
  const float* hrow = Hm + (size_t)n * HD + hh * D;
  const float* av = as_ + hh * D;
  const float* bv = ad_ + hh * D;
  float s = 0.f, d2 = 0.f;
  for (int d = 0; d < D; ++d) {
    float hv = hrow[d];
    s  += hv * av[d];
    d2 += hv * bv[d];
  }
  ES[i] = s;
  ED[i] = d2;
}

// ------------------------------ edge passes --------------------------------
__device__ __forceinline__ float edge_logit(const float* ES, const float* ED,
                                            int s, int d, int hh, int H) {
  float ev = ES[s * H + hh] + ED[d * H + hh];
  return ev > 0.f ? ev : 0.2f * ev;        // leaky_relu(0.2)
}

__global__ void edge_max_kernel(const int* __restrict__ src,
                                const int* __restrict__ dst,
                                const float* __restrict__ ES,
                                const float* __restrict__ ED,
                                unsigned int* __restrict__ MU, int E, int H) {
  int t = blockIdx.x * blockDim.x + threadIdx.x;
  if (t >= E * H) return;
  int e = t / H, hh = t - e * H;
  int s = src[e], d = dst[e];
  float ev = edge_logit(ES, ED, s, d, hh, H);
  atomicMax(&MU[d * H + hh], flip_f32(ev));
}

__global__ void edge_expsum_kernel(const int* __restrict__ src,
                                   const int* __restrict__ dst,
                                   const float* __restrict__ ES,
                                   const float* __restrict__ ED,
                                   const unsigned int* __restrict__ MU,
                                   float* __restrict__ DEN, int E, int H) {
  int t = blockIdx.x * blockDim.x + threadIdx.x;
  if (t >= E * H) return;
  int e = t / H, hh = t - e * H;
  int s = src[e], d = dst[e];
  float ev = edge_logit(ES, ED, s, d, hh, H);
  float m = unflip_f32(MU[d * H + hh]);
  atomicAdd(&DEN[d * H + hh], __expf(ev - m));
}

__global__ void edge_alpha_kernel(const int* __restrict__ src,
                                  const int* __restrict__ dst,
                                  const float* __restrict__ ES,
                                  const float* __restrict__ ED,
                                  const unsigned int* __restrict__ MU,
                                  const float* __restrict__ DEN,
                                  float* __restrict__ ALPHA, int E, int H) {
  int t = blockIdx.x * blockDim.x + threadIdx.x;
  if (t >= E * H) return;
  int e = t / H, hh = t - e * H;
  int s = src[e], d = dst[e];
  float ev = edge_logit(ES, ED, s, d, hh, H);
  float m = unflip_f32(MU[d * H + hh]);
  ALPHA[t] = __expf(ev - m) / (DEN[d * H + hh] + 1e-16f);
}

// One wave per edge; lane covers 4 consecutive features (float4 gather).
// outMask folds heads for the final layer (out row width = outMask+1).
__global__ __launch_bounds__(256) void edge_aggregate_kernel(
    const int* __restrict__ src, const int* __restrict__ dst,
    const float* __restrict__ ALPHA, const float* __restrict__ Hm,
    float* __restrict__ out, int E, int H, int HD, int log2D, int outMask,
    float scale) {
  int e = (blockIdx.x << 3) + (threadIdx.x >> 5);
  if (e >= E) return;
  int lane = threadIdx.x & 31;
  int s = src[e], d = dst[e];
  const float* hrow = Hm + (size_t)s * HD;
  float* orow = out + (size_t)d * (outMask + 1);
  for (int f0 = lane * 4; f0 < HD; f0 += 128) {
    float4 hv = *(const float4*)(hrow + f0);
    float al = ALPHA[e * H + (f0 >> log2D)] * scale;
    int ob = f0 & outMask;
    atomicAdd(orow + ob + 0, hv.x * al);
    atomicAdd(orow + ob + 1, hv.y * al);
    atomicAdd(orow + ob + 2, hv.z * al);
    atomicAdd(orow + ob + 3, hv.w * al);
  }
}

__global__ void elu_bf16_kernel(const float* __restrict__ in,
                                unsigned short* __restrict__ out, int n) {
  int i = blockIdx.x * blockDim.x + threadIdx.x;
  if (i >= n) return;
  float v = in[i];
  float r = v > 0.f ? v : (__expf(v) - 1.f);
  out[i] = f2bf(r);
}

// ------------------------------- launcher ----------------------------------

extern "C" void kernel_launch(void* const* d_in, const int* in_sizes, int n_in,
                              void* d_out, int out_size, void* d_ws,
                              size_t ws_size, hipStream_t stream) {
  const float* x   = (const float*)d_in[0];
  const int*   src = (const int*)d_in[1];
  const int*   dst = (const int*)d_in[2];
  const float* W1  = (const float*)d_in[3];
  const float* a1s = (const float*)d_in[4];
  const float* a1d = (const float*)d_in[5];
  const float* W2  = (const float*)d_in[6];
  const float* a2s = (const float*)d_in[7];
  const float* a2d = (const float*)d_in[8];
  const float* W3  = (const float*)d_in[9];
  const float* a3s = (const float*)d_in[10];
  const float* a3d = (const float*)d_in[11];

  const int N = in_sizes[0] / 96;
  const int E = in_sizes[1];
  const int H = 4;
  (void)n_in; (void)out_size; (void)ws_size;

  // ---- carve workspace (all buffers fully initialized every call) ----
  char* ws = (char*)d_ws;
  size_t off = 0;
  auto carve = [&](size_t bytes) -> void* {
    void* p = ws + off;
    off = (off + bytes + 255) & ~(size_t)255;
    return p;
  };
  unsigned short* Abf = (unsigned short*)carve((size_t)N * 128 * 2);
  unsigned short* W1t = (unsigned short*)carve((size_t)96  * 128 * 2);
  unsigned short* W2t = (unsigned short*)carve((size_t)128 * 128 * 2);
  unsigned short* W3t = (unsigned short*)carve((size_t)128 * 512 * 2);
  float*        Hbuf  = (float*)carve((size_t)N * 512 * 4);
  float*        AGG   = (float*)carve((size_t)N * 128 * 4);
  float*        ES    = (float*)carve((size_t)N * H * 4);
  float*        ED    = (float*)carve((size_t)N * H * 4);
  unsigned int* MU    = (unsigned int*)carve((size_t)N * H * 4);
  float*        DEN   = (float*)carve((size_t)N * H * 4);
  float*        ALPHA = (float*)carve((size_t)E * H * 4);

  const int TB = 256;
  auto grid1 = [&](long long n) {
    return dim3((unsigned)((n + TB - 1) / TB));
  };

  // ---- weight prep (bf16, transposed for WMMA B-fragment dword loads) ----
  wt_convert_kernel<<<grid1(96  * 128), TB, 0, stream>>>(W1, W1t, 96, 128);
  wt_convert_kernel<<<grid1(128 * 128), TB, 0, stream>>>(W2, W2t, 128, 128);
  wt_convert_kernel<<<grid1(128 * 512), TB, 0, stream>>>(W3, W3t, 128, 512);
  f32_to_bf16_kernel<<<grid1((long long)N * 96), TB, 0, stream>>>(x, Abf, N * 96);

  const unsigned int MINF = 0x007FFFFFu;  // flip_f32(-inf)

  struct Layer {
    int K; const unsigned short* Wt; const float* as; const float* ad; int D;
  };
  Layer layers[3] = {
      { 96, W1t, a1s, a1d, 32},
      {128, W2t, a2s, a2d, 32},
      {128, W3t, a3s, a3d, 128},
  };

  for (int li = 0; li < 3; ++li) {
    const Layer& L = layers[li];
    const int HD = H * L.D;

    // 1) H = elu-input @ W  (bf16 WMMA, f32 accumulate, LDS-staged B)
    const int mTiles  = (N + 15) / 16;
    const int mGroups = (mTiles + 7) / 8;
    const int nTiles  = HD / 16;
    gemm_wmma_bf16<<<dim3(nTiles * mGroups), 256, 0, stream>>>(
        Abf, L.Wt, Hbuf, N, L.K, HD);

    // 2) per-node attention scores
    att_scores_kernel<<<grid1((long long)N * H), TB, 0, stream>>>(
        Hbuf, L.as, L.ad, ES, ED, N, H, L.D);

    // 3) init reduction buffers
    fill_u32_kernel<<<grid1((long long)N * H), TB, 0, stream>>>(
        MU, MINF, (long long)N * H);
    fill_u32_kernel<<<grid1((long long)N * H), TB, 0, stream>>>(
        (unsigned int*)DEN, 0u, (long long)N * H);
    float* outbuf = (li == 2) ? (float*)d_out : AGG;
    fill_u32_kernel<<<grid1((long long)N * 128), TB, 0, stream>>>(
        (unsigned int*)outbuf, 0u, (long long)N * 128);

    // 4) segment softmax (max / exp-sum / alpha) + weighted aggregation
    edge_max_kernel<<<grid1((long long)E * H), TB, 0, stream>>>(
        src, dst, ES, ED, MU, E, H);
    edge_expsum_kernel<<<grid1((long long)E * H), TB, 0, stream>>>(
        src, dst, ES, ED, MU, DEN, E, H);
    edge_alpha_kernel<<<grid1((long long)E * H), TB, 0, stream>>>(
        src, dst, ES, ED, MU, DEN, ALPHA, E, H);

    const int log2D = (L.D == 32) ? 5 : 7;
    const float scale = (li == 2) ? 0.25f : 1.0f;  // fold head-mean into agg
    edge_aggregate_kernel<<<dim3((E + 7) / 8), 256, 0, stream>>>(
        src, dst, ALPHA, Hbuf, outbuf, E, H, HD, log2D, /*outMask=*/127, scale);

    // 5) ELU + bf16 convert feeds next layer's GEMM
    if (li < 2)
      elu_bf16_kernel<<<grid1((long long)N * HD), TB, 0, stream>>>(
          AGG, Abf, N * HD);
  }
}
